// MultiHeadAttention_32023276159708
// MI455X (gfx1250) — matrix operand order repaired and checked by reading, not compiled, not run on hardware
//
#include <hip/hip_runtime.h>
#include <hip/hip_bf16.h>

// ---------------- problem constants (match reference) ----------------
constexpr int S    = 2048;
constexpr int D    = 2048;
constexpr int H    = 16;
constexpr int HKV  = 8;
constexpr int HD   = 128;
constexpr int WIN  = 1024;
#define SCALE_F 0.08838834764831845f   // 128^-0.5
#define EPS_F   1e-6f

typedef __bf16 bf16_t;
typedef bf16_t v8bf  __attribute__((ext_vector_type(8)));
typedef bf16_t v16bf __attribute__((ext_vector_type(16)));
typedef float  v8f   __attribute__((ext_vector_type(8)));
typedef int    v4i   __attribute__((ext_vector_type(4)));

// ---------------- gfx1250 async global->LDS path ----------------
#if __has_builtin(__builtin_amdgcn_global_load_async_to_lds_b128)
#define ASYNC_LDS 1
#else
#define ASYNC_LDS 0
#endif

#if ASYNC_LDS
typedef __attribute__((address_space(1))) v4i* gas_ptr;   // global int4*
typedef __attribute__((address_space(3))) v4i* lds_ptr;   // LDS int4*
__device__ __forceinline__ void async_copy16(void* lds, const void* g) {
    // 16-byte global -> LDS DMA, tracked by ASYNCcnt (no VGPR round-trip)
    __builtin_amdgcn_global_load_async_to_lds_b128(
        (gas_ptr)(void*)g, (lds_ptr)lds, 0, 0);
}
__device__ __forceinline__ void async_wait0() {
#if __has_builtin(__builtin_amdgcn_s_wait_asynccnt)
    __builtin_amdgcn_s_wait_asynccnt(0);
#else
    asm volatile("s_wait_asynccnt 0x0" ::: "memory");
#endif
}
#endif

// Build a 16-element A/B fragment from two contiguous 8-element LDS/global reads.
__device__ __forceinline__ v16bf frag_from(const bf16_t* lo_p, const bf16_t* hi_p) {
    v8bf lo = *(const v8bf*)lo_p;
    v8bf hi = *(const v8bf*)hi_p;
    v16bf r;
#pragma unroll
    for (int j = 0; j < 8; ++j) { r[j] = lo[j]; r[j + 8] = hi[j]; }
    return r;
}

// =====================================================================
// Tiled GEMM:  C[M x N] = A[M x K] * B[K x N]
//   A: fp32 or bf16 (template), row-major.  B: fp32 row-major (weights).
//   C: fp32 or bf16 (template), row-major.
// Tile: BM=64, BN=128, BK=32. 256 threads = 8 waves (4 Mx, 2 Nx),
// each wave computes 16x64 via 4 accumulators of v_wmma_f32_16x16x32_bf16.
// =====================================================================
template <bool A_BF16, bool OUT_BF16>
__global__ __launch_bounds__(256) void gemm_wmma(const void* __restrict__ Av,
                                                 const float* __restrict__ B,
                                                 void* __restrict__ Cv,
                                                 int M, int N, int K) {
    constexpr int BM = 64, BN = 128, BK = 32;
    constexpr int LDA = 40, LDB = 40;               // +8 bf16 pad, keeps 16B alignment
    __shared__ alignas(16) bf16_t sA[BM * LDA];     // [m][k]
    __shared__ alignas(16) bf16_t sB[BN * LDB];     // [n][k]  (transposed)

    const int t    = threadIdx.x;
    const int lane = t & 31, wave = t >> 5;
    const int wm   = wave & 3, wn = wave >> 2;      // 4 x 2 wave grid
    const int ln   = lane & 15, hi = lane >> 4;
    const int m0   = blockIdx.y * BM, n0 = blockIdx.x * BN;

    v8f acc[4];
#pragma unroll
    for (int f = 0; f < 4; ++f)
#pragma unroll
        for (int r = 0; r < 8; ++r) acc[f][r] = 0.f;

    for (int k0 = 0; k0 < K; k0 += BK) {
        __syncthreads();
        // ---- stage A tile (convert fp32 -> bf16 if needed) ----
        if constexpr (A_BF16) {
            const bf16_t* A = (const bf16_t*)Av;
            int g = t;                               // 256 groups of 8 = 64x32
            int row = g >> 2, c8 = (g & 3) * 8;
#if ASYNC_LDS
            async_copy16(sA + row * LDA + c8,
                         A + (size_t)(m0 + row) * K + k0 + c8);
#else
            v8bf x = *(const v8bf*)(A + (size_t)(m0 + row) * K + k0 + c8);
            *(v8bf*)(sA + row * LDA + c8) = x;
#endif
        } else {
            const float* A = (const float*)Av;
#pragma unroll
            for (int i = 0; i < 2; ++i) {
                int g = t + i * 256;                 // 512 float4 = 64x32 fp32
                int row = g >> 3, c4 = (g & 7) * 4;
                float4 x = *(const float4*)(A + (size_t)(m0 + row) * K + k0 + c4);
                bf16_t* d = sA + row * LDA + c4;
                d[0] = (bf16_t)x.x; d[1] = (bf16_t)x.y;
                d[2] = (bf16_t)x.z; d[3] = (bf16_t)x.w;
            }
        }
        // ---- stage B tile transposed: sB[n][k], fp32 -> bf16 ----
#pragma unroll
        for (int i = 0; i < 4; ++i) {
            int g = t + i * 256;                     // 1024 float4 = 32x128 fp32
            int row = g >> 5, c4 = (g & 31) * 4;     // row = k index
            float4 x = *(const float4*)(B + (size_t)(k0 + row) * N + n0 + c4);
            sB[(c4 + 0) * LDB + row] = (bf16_t)x.x;
            sB[(c4 + 1) * LDB + row] = (bf16_t)x.y;
            sB[(c4 + 2) * LDB + row] = (bf16_t)x.z;
            sB[(c4 + 3) * LDB + row] = (bf16_t)x.w;
        }
        // prefetch next k-tile into cache (global_prefetch_b8)
        if (k0 + BK < K) {
            __builtin_prefetch(B + (size_t)(k0 + BK) * N + n0 + (t & 31) * 4, 0, 0);
            if constexpr (!A_BF16)
                __builtin_prefetch((const float*)Av + (size_t)(m0 + (t >> 2)) * K + k0 + BK, 0, 0);
        }
#if ASYNC_LDS
        if constexpr (A_BF16) async_wait0();        // own ASYNCcnt before barrier
#endif
        __syncthreads();

        // ---- A fragment (16x32): lane=row, k = 8*hi + j (j<8), 16+8*hi+(j-8)
        const bf16_t* pa = sA + (wm * 16 + ln) * LDA + 8 * hi;
        v16bf a = frag_from(pa, pa + 16);
        // ---- 4 B fragments + WMMAs
#pragma unroll
        for (int f = 0; f < 4; ++f) {
            const bf16_t* pb = sB + (wn * 64 + f * 16 + ln) * LDB + 8 * hi;
            v16bf b = frag_from(pb, pb + 16);
            acc[f] = __builtin_amdgcn_wmma_f32_16x16x32_bf16(
                false, a, false, b, (short)0, acc[f], false, false);
        }
    }

    // ---- write C (f32 16x16 layout: N = lane&15, M = 8*hi + r) ----
    const int mrow = m0 + wm * 16 + hi * 8;
    const int ncol = n0 + wn * 64 + ln;
#pragma unroll
    for (int f = 0; f < 4; ++f)
#pragma unroll
        for (int r = 0; r < 8; ++r) {
            size_t idx = (size_t)(mrow + r) * N + ncol + f * 16;
            if constexpr (OUT_BF16) ((bf16_t*)Cv)[idx] = (bf16_t)acc[f][r];
            else                    ((float*)Cv)[idx]  = acc[f][r];
        }
}

// =====================================================================
// Fused per-head RMSNorm + RoPE, fp32 in -> bf16 out.
// One block (128 threads) per (s, head) row of 128 dims.
// =====================================================================
__global__ __launch_bounds__(128) void rmsnorm_rope(const float* __restrict__ x,
                                                    bf16_t* __restrict__ y,
                                                    const float* __restrict__ scale,
                                                    const float* __restrict__ cosb,
                                                    const float* __restrict__ sinb,
                                                    int NH) {
    __shared__ float xs[128];
    __shared__ float part[4];
    const int row = blockIdx.x;          // s*NH + h
    const int s   = row / NH;
    const int d   = threadIdx.x;

    float v  = x[(size_t)row * HD + d];
    float ss = v * v;
#pragma unroll
    for (int o = 16; o >= 1; o >>= 1) ss += __shfl_xor(ss, o, 32);
    if ((d & 31) == 0) part[d >> 5] = ss;
    __syncthreads();
    float var = (part[0] + part[1] + part[2] + part[3]) * (1.f / 128.f);
    float rn  = rsqrtf(var + EPS_F);
    float xn  = v * rn * scale[d];
    xs[d] = xn;
    __syncthreads();
    float rot = (d < 64) ? -xs[d + 64] : xs[d - 64];
    float c   = cosb[(size_t)s * HD + d];
    float sn  = sinb[(size_t)s * HD + d];
    y[(size_t)row * HD + d] = (bf16_t)(xn * c + rot * sn);
}

// =====================================================================
// Flash attention with sliding window + GQA (g=2), all-bf16 operands,
// fp32 accumulation & softmax. 128 threads = 4 waves; each wave owns a
// 16-query tile; the workgroup shares K/V LDS tiles of 32 keys.
// K tile is staged via GLOBAL_LOAD_ASYNC_TO_LDS_B128 (ASYNCcnt) when
// available; V is transposed through VGPRs (layout change required).
// =====================================================================
__global__ __launch_bounds__(128) void flash_attn(const bf16_t* __restrict__ qg,
                                                  const bf16_t* __restrict__ kg,
                                                  const bf16_t* __restrict__ vg,
                                                  bf16_t* __restrict__ og) {
    constexpr int LK = 136, LV = 40, LP = 40;
    __shared__ alignas(16) bf16_t sK[32 * LK];       // [key][dim]
    __shared__ alignas(16) bf16_t sV[128 * LV];      // [dim][key]  (V^T)
    __shared__ alignas(16) bf16_t sP[4 * 16 * LP];   // per-wave P [m][key]

    const int t = threadIdx.x, lane = t & 31, w = t >> 5;
    const int ln = lane & 15, hi = lane >> 4;
    const int h = blockIdx.y, hk = h >> 1;
    const int q0 = blockIdx.x * 64;
    const int qrow = q0 + w * 16 + ln;

    // Q A-fragments (16 x 128 -> 4 chunks of 16x32), kept in registers
    v16bf qa[4];
#pragma unroll
    for (int c = 0; c < 4; ++c) {
        const bf16_t* p = qg + ((size_t)qrow * H + h) * HD + 32 * c + 8 * hi;
        qa[c] = frag_from(p, p + 16);
    }

    v8f acc[8];
    float mrow[8], lrow[8];
#pragma unroll
    for (int f = 0; f < 8; ++f)
#pragma unroll
        for (int r = 0; r < 8; ++r) acc[f][r] = 0.f;
#pragma unroll
    for (int r = 0; r < 8; ++r) { mrow[r] = -1e30f; lrow[r] = 0.f; }

    int kb_lo = q0 - WIN; if (kb_lo < 0) kb_lo = 0;
    const int kb_hi = q0 + 64;

    for (int kb = kb_lo; kb < kb_hi; kb += 32) {
        __syncthreads();
        // ---- cooperative stage: K tile async (byte-identical copy),
        //      V tile transposed through VGPRs ----
#pragma unroll
        for (int i = 0; i < 4; ++i) {
            int g = t + i * 128;                     // 512 groups of 8 = 32x128
            int row = g >> 4, c8 = (g & 15) * 8;
            size_t src = ((size_t)(kb + row) * HKV + hk) * HD + c8;
#if ASYNC_LDS
            async_copy16(sK + row * LK + c8, kg + src);
#else
            v8bf kv = *(const v8bf*)(kg + src);
            *(v8bf*)(sK + row * LK + c8) = kv;
#endif
            v8bf vv = *(const v8bf*)(vg + src);
#pragma unroll
            for (int j = 0; j < 8; ++j) sV[(c8 + j) * LV + row] = vv[j];
        }
#if ASYNC_LDS
        async_wait0();                               // own ASYNCcnt before barrier
#endif
        __syncthreads();

        // ---- S = Q K^T for 32 keys (two 16-col halves) ----
        v8f sc[2];
#pragma unroll
        for (int half = 0; half < 2; ++half) {
            v8f s;
#pragma unroll
            for (int r = 0; r < 8; ++r) s[r] = 0.f;
            const int key = half * 16 + ln;
#pragma unroll
            for (int c = 0; c < 4; ++c) {
                const bf16_t* p = sK + key * LK + 32 * c + 8 * hi;
                v16bf b = frag_from(p, p + 16);
                s = __builtin_amdgcn_wmma_f32_16x16x32_bf16(
                    false, qa[c], false, b, (short)0, s, false, false);
            }
            sc[half] = s;
        }

        // ---- online softmax (row stats per half-wave) ----
        float pm[2][8];
#pragma unroll
        for (int r = 0; r < 8; ++r) {
            const int i = q0 + w * 16 + hi * 8 + r;
            float best = -1e30f;
#pragma unroll
            for (int half = 0; half < 2; ++half) {
                int j = kb + half * 16 + ln;
                bool ok = (j <= i) && (i - j <= WIN);
                float val = ok ? sc[half][r] * SCALE_F : -1e30f;
                pm[half][r] = val;
                best = fmaxf(best, val);
            }
#pragma unroll
            for (int o = 8; o >= 1; o >>= 1) best = fmaxf(best, __shfl_xor(best, o, 32));
            float mn   = fmaxf(mrow[r], best);
            float corr = __expf(mrow[r] - mn);
            mrow[r] = mn;
            float psum = 0.f;
#pragma unroll
            for (int half = 0; half < 2; ++half) {
                float p = (pm[half][r] > -9e29f) ? __expf(pm[half][r] - mn) : 0.f;
                sP[(w * 16 + hi * 8 + r) * LP + half * 16 + ln] = (bf16_t)p;
                psum += p;
            }
#pragma unroll
            for (int o = 8; o >= 1; o >>= 1) psum += __shfl_xor(psum, o, 32);
            lrow[r] = lrow[r] * corr + psum;
#pragma unroll
            for (int f = 0; f < 8; ++f) acc[f][r] *= corr;
        }

        // ---- O += P V : re-fragment P via LDS, 8 WMMAs over dim chunks ----
        const bf16_t* pp = sP + (w * 16 + ln) * LP + 8 * hi;
        v16bf pa = frag_from(pp, pp + 16);
#pragma unroll
        for (int f = 0; f < 8; ++f) {
            const bf16_t* pb = sV + (f * 16 + ln) * LV + 8 * hi;
            v16bf b = frag_from(pb, pb + 16);
            acc[f] = __builtin_amdgcn_wmma_f32_16x16x32_bf16(
                false, pa, false, b, (short)0, acc[f], false, false);
        }
    }

    // ---- epilogue: normalize and emit bf16 [s][h*128 + d] ----
#pragma unroll
    for (int f = 0; f < 8; ++f)
#pragma unroll
        for (int r = 0; r < 8; ++r) {
            int m = hi * 8 + r;
            float ov = acc[f][r] / lrow[r];
            og[((size_t)(q0 + w * 16 + m) * H + h) * HD + f * 16 + ln] = (bf16_t)ov;
        }
}

// =====================================================================
// Launch
// =====================================================================
extern "C" void kernel_launch(void* const* d_in, const int* in_sizes, int n_in,
                              void* d_out, int out_size, void* d_ws, size_t ws_size,
                              hipStream_t stream) {
    (void)in_sizes; (void)n_in; (void)out_size; (void)ws_size;
    const float* X    = (const float*)d_in[0];
    const float* cosb = (const float*)d_in[1];
    const float* sinb = (const float*)d_in[2];
    const float* Wq   = (const float*)d_in[3];
    const float* Wk   = (const float*)d_in[4];
    const float* Wv   = (const float*)d_in[5];
    const float* Wo   = (const float*)d_in[6];
    const float* qsc  = (const float*)d_in[7];
    const float* ksc  = (const float*)d_in[8];
    float* out = (float*)d_out;

    char* ws = (char*)d_ws;
    size_t off = 0;
    float*  qf = (float*)(ws + off);  off += (size_t)S * (H  * HD) * 4;  // 16 MB
    float*  kf = (float*)(ws + off);  off += (size_t)S * (HKV * HD) * 4; //  8 MB
    bf16_t* qb = (bf16_t*)(ws + off); off += (size_t)S * (H  * HD) * 2;  //  8 MB
    bf16_t* kb = (bf16_t*)(ws + off); off += (size_t)S * (HKV * HD) * 2; //  4 MB
    bf16_t* vb = (bf16_t*)(ws + off); off += (size_t)S * (HKV * HD) * 2; //  4 MB
    bf16_t* ab = (bf16_t*)(ws + off); off += (size_t)S * (H  * HD) * 2;  //  8 MB

    // 1-3: QKV projections (V written directly as bf16; Q/K as fp32 for norm)
    gemm_wmma<false, false><<<dim3((H  * HD) / 128, S / 64), 256, 0, stream>>>(X, Wq, qf, S, H  * HD, D);
    gemm_wmma<false, false><<<dim3((HKV * HD) / 128, S / 64), 256, 0, stream>>>(X, Wk, kf, S, HKV * HD, D);
    gemm_wmma<false, true ><<<dim3((HKV * HD) / 128, S / 64), 256, 0, stream>>>(X, Wv, vb, S, HKV * HD, D);
    // 4-5: RMSNorm + RoPE -> bf16
    rmsnorm_rope<<<S * H,   128, 0, stream>>>(qf, qb, qsc, cosb, sinb, H);
    rmsnorm_rope<<<S * HKV, 128, 0, stream>>>(kf, kb, ksc, cosb, sinb, HKV);
    // 6: sliding-window GQA flash attention
    flash_attn<<<dim3(S / 64, H), 128, 0, stream>>>(qb, kb, vb, ab);
    // 7: output projection -> fp32 d_out
    gemm_wmma<true, false><<<dim3(D / 128, S / 64), 256, 0, stream>>>(ab, Wo, out, S, D, H * HD);
}
